// Int4EinsumBias_40913858462383
// MI455X (gfx1250) — matrix operand order repaired and checked by reading, not compiled, not run on hardware
//
#include <hip/hip_runtime.h>

// ---------------------------------------------------------------------------
// Int4 (stored as int32) grouped-dequant fused QKV projection, bf16 WMMA.
//   out[bt, n, m] = sum_k x[bt,k] * (q[n,k,m] - z[n,k/G,m]) * s[n,k/G,m]
//   out tail     = bias (pass-through, per reference tuple)
//
// CDNA5 paths used:
//   - v_wmma_f32_16x16x32_bf16 (16 per k-stage per wave)
//   - TENSOR_LOAD_TO_LDS (TDM DMA of the x tile) + s_wait_tensorcnt
//   - global_store_async_from_lds_b128 + s_wait_asynccnt (C epilogue)
//   - global_prefetch_b8 (qweight tile prefetch into GL2)
//   - double-buffered LDS pipeline
// ---------------------------------------------------------------------------

typedef __attribute__((ext_vector_type(16))) __bf16 bf16x16;
typedef __attribute__((ext_vector_type(8)))  __bf16 bf16x8;
typedef __attribute__((ext_vector_type(8)))  float  f32x8;
typedef __attribute__((ext_vector_type(4)))  unsigned int u32x4;
typedef __attribute__((ext_vector_type(4)))  int          i32x4;
typedef __attribute__((ext_vector_type(8)))  int          i32x8;

union FragU { bf16x16 v; bf16x8 h[2]; };

// Problem constants (from reference setup_inputs)
constexpr int BT    = 8192;   // B*T
constexpr int K     = 4096;
constexpr int M     = 4096;
constexpr int NPROJ = 3;
constexpr int G     = 128;    // quant group size

// Tiling
constexpr int BM = 128;       // block tile rows (BT dim)
constexpr int BN = 128;       // block tile cols (M dim)
constexpr int BK = 64;        // k step per stage (fits inside one quant group)
constexpr int KT = K / BK;    // 64 k-stages
constexpr int TILE = BM * BK; // elements per LDS tile (16 KB as bf16)

__global__ __launch_bounds__(256)
void int4_qkv_wmma_kernel(const __bf16* __restrict__ x,
                          const __bf16* __restrict__ scales,
                          const int*    __restrict__ qweight,
                          const int*    __restrict__ qzeros,
                          __bf16*       __restrict__ out)
{
    // [A0 16K][A1 16K][B0 16K][B1 16K]; C tile (32K) reuses the A half.
    __shared__ __align__(16) __bf16 smem[4 * TILE];

    const int tid   = threadIdx.x;
    const int lane  = tid & 31;
    const int wave  = tid >> 5;
    const int wr    = wave & 1;     // 0..1 : 64-row group
    const int wc    = wave >> 1;    // 0..3 : 32-col group
    const int khalf = lane >> 4;    // 0/1  : wave32 half, per WMMA VGPR layout
    const int lrow  = lane & 15;

    const int bt0  = blockIdx.x * BM;
    const int m0   = blockIdx.y * BN;
    const int proj = blockIdx.z;

    const int*    qbase = qweight + (size_t)proj * K * M;
    const int*    zbase = qzeros  + (size_t)proj * (K / G) * M;
    const __bf16* sbase = scales  + (size_t)proj * (K / G) * M;

    // LDS buffer selectors: computed pointers only (no pointer aggregates —
    // those constant-fold into addrspacecast static initializers and break lld).
    auto ldsA  = [&](int buf) -> __bf16* { return smem + buf * TILE; };
    auto ldsBt = [&](int buf) -> __bf16* { return smem + (2 + buf) * TILE; };
    __bf16* ldsC = smem;                      // 128x128 bf16, post-loop only

    // ---- staging helpers -------------------------------------------------
    // x tile: one TDM descriptor moves the whole 128x64 bf16 tile global->LDS.
    // Issued by wave 0 only; TENSORcnt fenced after the WMMA block.
    auto stage_a_tdm = [&](int kt, int buf) {
        if (wave != 0) return;
        __bf16* dstA = ldsA(buf);
        const int k0 = kt * BK;
        const unsigned long long ga =
            (unsigned long long)(size_t)(const void*)
                (x + (size_t)bt0 * K + k0);       // tile start, byte address

        u32x4 g0;                                  // D# group 0 (128b)
        g0.x = 0x1u;                               // count=1, user mode
        g0.y = (unsigned)(size_t)dstA;             // lds_addr (LDS byte offset)
        g0.z = (unsigned)(ga & 0xFFFFFFFFu);       // global_addr[31:0]
        g0.w = (unsigned)((ga >> 32) & 0x01FFFFFFu)
             | (2u << 30);                         // global_addr[56:32] | type=2

        i32x8 g1;                                  // D# group 1 (256b)
        g1[0] = 1 << 16;                           // data_size=1 (2B), mask=0
        g1[1] = (K & 0xFFFF) << 16;                // tensor_dim0[15:0]
        g1[2] = ((K >> 16) & 0xFFFF)               // tensor_dim0[31:16]
              | ((BT & 0xFFFF) << 16);             // tensor_dim1[15:0]
        g1[3] = ((BT >> 16) & 0xFFFF)              // tensor_dim1[31:16]
              | (BK << 16);                        // tile_dim0 = 64
        g1[4] = BM;                                // tile_dim1 = 128 (dim2 = 0)
        g1[5] = K;                                 // tensor_dim0_stride[31:0]
        g1[6] = 0;                                 // stride hi / dim1_stride lo
        g1[7] = 0;

        const i32x4 gz4 = {};                      // groups 2/3 unused (2D)
        const i32x8 gz8 = {};
        __builtin_amdgcn_tensor_load_to_lds(g0, g1, gz4, gz4, gz8, 0);
    };

    // weight tile: load int32 q, grouped dequant, store bf16 TRANSPOSED [m][k].
    // Each thread owns 4 fixed m-columns (one z/s fetch) and 8 k-rows.
    auto stage_b_dequant = [&](int kt, int buf) {
        __bf16* dstB = ldsBt(buf);
        const int k0  = kt * BK;
        const int g   = k0 / G;                  // BK=64 never crosses a group
        const int mq  = lane << 2;               // 0,4,...,124 (fixed per thread)
        const int klb = wave << 3;               // k-row base 0,8,...,56
        const int4 z = *(const int4*)(zbase + (size_t)g * M + m0 + mq);
        const __bf16* sp = sbase + (size_t)g * M + m0 + mq;
        const float s0 = (float)sp[0], s1 = (float)sp[1];
        const float s2 = (float)sp[2], s3 = (float)sp[3];
        #pragma unroll
        for (int p = 0; p < 8; ++p) {
            const int kl = klb + p;
            const int4 q = *(const int4*)(qbase + (size_t)(k0 + kl) * M + m0 + mq);
            dstB[(mq + 0) * BK + kl] = (__bf16)((float)(q.x - z.x) * s0);
            dstB[(mq + 1) * BK + kl] = (__bf16)((float)(q.y - z.y) * s1);
            dstB[(mq + 2) * BK + kl] = (__bf16)((float)(q.z - z.z) * s2);
            dstB[(mq + 3) * BK + kl] = (__bf16)((float)(q.w - z.w) * s3);
        }
    };

    auto prefetch_q = [&](int kt) {              // pull next q tile into GL2
        const int k0 = kt * BK;
        const int* nxt = qbase + (size_t)(k0 + (tid >> 2)) * M
                               + m0 + (tid & 3) * 32;
        __builtin_prefetch(nxt, 0, 1);
    };

    // ---- prologue: stage tile 0 into buffer 0 ----------------------------
    f32x8 acc[4][2] = {};
    stage_a_tdm(0, 0);
    prefetch_q(0);
    stage_b_dequant(0, 0);
    if (wave == 0) __builtin_amdgcn_s_wait_tensorcnt(0);
    __syncthreads();

    // ---- main pipeline ---------------------------------------------------
    for (int kt = 0; kt < KT; ++kt) {
        const int cur = kt & 1;
        const int nxt = cur ^ 1;
        const bool more = (kt + 1) < KT;

        if (more) {
            stage_a_tdm(kt + 1, nxt);            // TDM fire-and-forget
            prefetch_q(kt + 1);                  // GL2 warm-up for dequant below
        }

        // 2 WMMA k-steps of 32 over the staged BK=64
        const __bf16* As = ldsA(cur);
        const __bf16* Bs = ldsBt(cur);
        #pragma unroll
        for (int kk = 0; kk < BK; kk += 32) {
            bf16x16 a[4];
            #pragma unroll
            for (int i = 0; i < 4; ++i) {
                const int row = wr * 64 + i * 16 + lrow;
                FragU u;   // 16-bit A 16x32: lane half holds K chunks {h*8, 16+h*8}
                u.h[0] = *(const bf16x8*)&As[row * BK + kk + khalf * 8];
                u.h[1] = *(const bf16x8*)&As[row * BK + kk + 16 + khalf * 8];
                a[i] = u.v;
            }
            bf16x16 b[2];
            #pragma unroll
            for (int j = 0; j < 2; ++j) {
                const int col = wc * 32 + j * 16 + lrow;
                FragU u;   // 16-bit B 32x16: lane half holds K chunks {h*8, 16+h*8}
                u.h[0] = *(const bf16x8*)&Bs[col * BK + kk + khalf * 8];
                u.h[1] = *(const bf16x8*)&Bs[col * BK + kk + 16 + khalf * 8];
                b[j] = u.v;
            }
            #pragma unroll
            for (int i = 0; i < 4; ++i)
                #pragma unroll
                for (int j = 0; j < 2; ++j)
                    acc[i][j] = __builtin_amdgcn_wmma_f32_16x16x32_bf16(
                        false, a[i], false, b[j],
                        (short)0, acc[i][j], false, false);
        }

        if (more) {
            stage_b_dequant(kt + 1, nxt);        // q loads mostly hit GL2 now
            if (wave == 0) __builtin_amdgcn_s_wait_tensorcnt(0);
        }
        __syncthreads();
    }

    // ---- epilogue: C through LDS, then async LDS->global b128 stores -----
    const int radd = khalf * 8;                  // lanes 16-31 hold rows r+8
    #pragma unroll
    for (int i = 0; i < 4; ++i) {
        #pragma unroll
        for (int j = 0; j < 2; ++j) {
            const int col = wc * 32 + j * 16 + lrow;
            #pragma unroll
            for (int r = 0; r < 8; ++r) {
                const int row = wr * 64 + i * 16 + r + radd;
                ldsC[row * BN + col] = (__bf16)acc[i][j][r];
            }
        }
    }
    __syncthreads();
    #pragma unroll
    for (int p = 0; p < 8; ++p) {
        const int idx = p * 256 + tid;           // 2048 chunks of 8 bf16
        const int row = idx >> 4;
        const int c8  = (idx & 15) * 8;
        const unsigned lds_off = (unsigned)(size_t)&ldsC[row * BN + c8];
        const unsigned g_off = (unsigned)(
            (((size_t)(bt0 + row) * NPROJ + proj) * M + m0 + c8)
            * sizeof(__bf16));
        asm volatile("global_store_async_from_lds_b128 %0, %1, %2 offset:0"
                     :
                     : "v"(g_off), "v"(lds_off), "s"(out)
                     : "memory");
    }
    asm volatile("s_wait_asynccnt 0x0" ::: "memory");
}

__global__ void bias_copy_kernel(const __bf16* __restrict__ bias,
                                 __bf16* __restrict__ out_tail)
{
    const int i = blockIdx.x * blockDim.x + threadIdx.x;
    if (i < NPROJ * M) out_tail[i] = bias[i];
}

extern "C" void kernel_launch(void* const* d_in, const int* in_sizes, int n_in,
                              void* d_out, int out_size, void* d_ws, size_t ws_size,
                              hipStream_t stream)
{
    // setup_inputs order: x, scales, bias, qweight, qzeros, group_size
    const __bf16* x       = (const __bf16*)d_in[0];
    const __bf16* scales  = (const __bf16*)d_in[1];
    const __bf16* bias    = (const __bf16*)d_in[2];
    const int*    qweight = (const int*)d_in[3];
    const int*    qzeros  = (const int*)d_in[4];

    __bf16* out = (__bf16*)d_out;

    dim3 grid(BT / BM, M / BN, NPROJ);   // (64, 32, 3)
    dim3 block(256);
    int4_qkv_wmma_kernel<<<grid, block, 0, stream>>>(x, scales, qweight, qzeros, out);

    // bias is the 2nd tuple output, concatenated after the result
    __bf16* tail = out + (size_t)BT * NPROJ * M;
    bias_copy_kernel<<<(NPROJ * M + 255) / 256, 256, 0, stream>>>(bias, tail);
}
